// GraphSAG_32083405701297
// MI455X (gfx1250) — matrix operand order repaired and checked by reading, not compiled, not run on hardware
//
#include <hip/hip_runtime.h>
#include <hip/hip_bf16.h>
#include <math.h>

typedef _Float16 h8   __attribute__((ext_vector_type(8)));
typedef _Float16 v16h __attribute__((ext_vector_type(16)));
typedef float    v8f  __attribute__((ext_vector_type(8)));

// ---- K1: hw[j] = dot(h[j, 0:32768], W_top)  (memory-bound, vectorized) ----
__global__ void k_rowdot(const float* __restrict__ h, const float* __restrict__ w,
                         float* __restrict__ hw) {
  const int j = blockIdx.x;
  const int t = threadIdx.x;
  const float4* hp = reinterpret_cast<const float4*>(h + (size_t)j * 32768);
  const float4* wp = reinterpret_cast<const float4*>(w);
  float acc = 0.f;
#pragma unroll
  for (int k = 0; k < 32; ++k) {
    const int i = k * 256 + t;
    float4 a = hp[i], b = wp[i];
    acc += a.x * b.x + a.y * b.y + a.z * b.z + a.w * b.w;
  }
  __shared__ float red[256];
  red[t] = acc;
  __syncthreads();
  for (int s = 128; s > 0; s >>= 1) {
    if (t < s) red[t] += red[t + s];
    __syncthreads();
  }
  if (t == 0) hw[j] = red[0];
}

// ---- Generic matvec + activation: y[i] = act(alpha * (G[i,:]·x) + beta) ----
// mode 0: relu ; mode 1: sigmoid(relu(.))
__global__ void k_matvec_act(const float* __restrict__ G, const float* __restrict__ x,
                             int n, const float* __restrict__ alpha,
                             const float* __restrict__ beta, int mode,
                             float* __restrict__ out1, float* __restrict__ out2) {
  const int i = blockIdx.x;
  const int t = threadIdx.x;
  const float* row = G + (size_t)i * n;
  float acc = 0.f;
  for (int j = t; j < n; j += 256) acc += row[j] * x[j];
  __shared__ float red[256];
  red[t] = acc;
  __syncthreads();
  for (int s = 128; s > 0; s >>= 1) {
    if (t < s) red[t] += red[t + s];
    __syncthreads();
  }
  if (t == 0) {
    float a = alpha ? alpha[0] : 1.f;
    float y = fmaxf(a * red[0] + beta[0], 0.f);
    if (mode == 1) y = 1.f / (1.f + expf(-y));
    out1[i] = y;
    if (out2) out2[i] = y;
  }
}

// ---- Top-k (lax.top_k semantics): rank by (score desc, index asc) ----
__global__ void k_topk(const float* __restrict__ scores, const float* __restrict__ hcur,
                       int n, int kk, int* __restrict__ idx, float* __restrict__ hpool) {
  __shared__ float s[512];
  const int t = threadIdx.x;
  s[t] = (t < n) ? scores[t] : -1.f;
  __syncthreads();
  if (t < n) {
    const float si = s[t];
    int rank = 0;
    for (int j = 0; j < n; ++j) {
      const float sj = s[j];
      rank += (sj > si) || (sj == si && j < t);
    }
    if (rank < kk) {
      idx[rank] = t;
      hpool[rank] = hcur[t] * si;  // new_h = h[idx] * values
    }
  }
}

// ---- Binarize current graph into padded f16 matrix + its transpose ----
__global__ void k_binpad(const float* __restrict__ g, int n, int P,
                         _Float16* __restrict__ Gh, _Float16* __restrict__ GTh) {
  const int tid = blockIdx.x * 256 + threadIdx.x;
  if (tid >= P * P) return;
  const int i = tid / P, j = tid % P;
  _Float16 b = (_Float16)0;
  if (i < n && j < n && g[(size_t)i * n + j] != 0.f) b = (_Float16)1;
  Gh[(size_t)i * P + j]  = b;
  GTh[(size_t)j * P + i] = b;
}

// ---- 2-hop connectivity via WMMA: B2 = (Gh @ Gh != 0), one wave per 16x16 tile ----
__global__ void k_twohop_wmma(const _Float16* __restrict__ Gh,
                              const _Float16* __restrict__ GTh,
                              int P, float* __restrict__ B2) {
  const int lane = threadIdx.x;          // 32-thread block -> EXEC all ones
  const int tn = blockIdx.x, tm = blockIdx.y;
  const int hi  = (lane >= 16) ? 1 : 0;
  const int l15 = lane & 15;
  const h8* arow = reinterpret_cast<const h8*>(Gh  + (size_t)(tm * 16 + l15) * P);
  const h8* brow = reinterpret_cast<const h8*>(GTh + (size_t)(tn * 16 + l15) * P);
  v8f acc = {};
  const int KT = P >> 5;
  for (int kt = 0; kt < KT; ++kt) {
    // A 16x32 f16 layout: lanes<16 K={0..7,16..23}, lanes>=16 K={8..15,24..31}
    h8 alo = arow[kt * 4 + hi];
    h8 ahi = arow[kt * 4 + 2 + hi];
    // B 32x16 f16 layout: lane = column N, lanes<16 K={0..7,16..23}, lanes>=16 K={8..15,24..31}
    h8 blo = brow[kt * 4 + hi];
    h8 bhi = brow[kt * 4 + 2 + hi];
    v16h a, b;
#pragma unroll
    for (int e = 0; e < 8; ++e) {
      a[e] = alo[e]; a[e + 8] = ahi[e];
      b[e] = blo[e]; b[e + 8] = bhi[e];
    }
    acc = __builtin_amdgcn_wmma_f32_16x16x32_f16(false, a, false, b,
                                                 (short)0, acc, false, false);
  }
  // C/D: VGPR r -> row tm*16 + r + 8*(lane/16), col tn*16 + lane%16
  const int r0 = tm * 16 + 8 * hi;
  const int c  = tn * 16 + l15;
#pragma unroll
  for (int r = 0; r < 8; ++r)
    B2[(size_t)(r0 + r) * P + c] = (acc[r] != 0.f) ? 1.f : 0.f;
}

// ---- Row sums of the gathered submatrix: S[j] = sum_k B2[idx[j], idx[k]] ----
__global__ void k_rowsum(const float* __restrict__ B2, int P,
                         const int* __restrict__ idx, int kk, float* __restrict__ S) {
  __shared__ int id[512];
  const int t = threadIdx.x;
  if (t < kk) id[t] = idx[t];
  __syncthreads();
  if (t < kk) {
    const float* row = B2 + (size_t)id[t] * P;
    float s = 0.f;
    for (int k = 0; k < kk; ++k) s += row[id[k]];
    S[t] = s;
  }
}

// ---- g_next[i,j] = B2[idx[i], idx[j]] / S[j]  (column-normalize quirk kept) ----
__global__ void k_gathernorm(const float* __restrict__ B2, int P,
                             const int* __restrict__ idx, const float* __restrict__ S,
                             int kk, float* __restrict__ gn) {
  const int t = blockIdx.x * 256 + threadIdx.x;
  if (t >= kk * kk) return;
  const int i = t / kk, j = t % kk;
  gn[t] = B2[(size_t)idx[i] * P + idx[j]] / S[j];
}

// ---- xd[j] = h_pool[j] * Wd[j]  (folds diag GCN into a matvec) ----
__global__ void k_prescale(const float* __restrict__ hpool, const float* __restrict__ Wd,
                           int kk, float* __restrict__ xd) {
  const int t = blockIdx.x * 256 + threadIdx.x;
  if (t < kk) xd[t] = hpool[t] * Wd[t];
}

extern "C" void kernel_launch(void* const* d_in, const int* in_sizes, int n_in,
                              void* d_out, int out_size, void* d_ws, size_t ws_size,
                              hipStream_t stream) {
  (void)in_sizes; (void)n_in; (void)out_size; (void)ws_size;
  const float* g  = (const float*)d_in[0];
  const float* h  = (const float*)d_in[1];
  const float* Wt = (const float*)d_in[2];
  const float* bt = (const float*)d_in[3];
  float* out = (float*)d_out;

  char* w = (char*)d_ws;
  auto alloc = [&](size_t bytes) -> void* {
    void* p = (void*)w;
    w += (bytes + 255) & ~(size_t)255;
    return p;
  };
  float*    hw     = (float*)alloc(512 * 4);
  float*    hcur   = (float*)alloc(512 * 4);
  float*    scores = (float*)alloc(512 * 4);
  float*    hpool  = (float*)alloc(512 * 4);
  float*    xd     = (float*)alloc(512 * 4);
  float*    S      = (float*)alloc(512 * 4);
  int*      idx    = (int*)  alloc(512 * 4);
  float*    gA     = (float*)alloc((size_t)512 * 512 * 4);
  float*    gB     = (float*)alloc((size_t)512 * 512 * 4);
  float*    B2     = (float*)alloc((size_t)512 * 512 * 4);
  _Float16* Gh     = (_Float16*)alloc((size_t)512 * 512 * 2);
  _Float16* GTh    = (_Float16*)alloc((size_t)512 * 512 * 2);

  // Top GCN, reassociated: ht = relu(g @ (h @ W_top^T) + b_top)  -> out[0:500]
  k_rowdot<<<500, 256, 0, stream>>>(h, Wt, hw);
  k_matvec_act<<<500, 256, 0, stream>>>(g, hw, 500, nullptr, bt, 0, hcur, out);

  const int ns[4]   = {500, 400, 300, 200};
  const int kks[4]  = {400, 300, 200, 100};  // max(2, int(k*n)) precomputed
  const int offs[4] = {500, 900, 1200, 1400};
  const float* gcur = g;
  for (int s = 0; s < 4; ++s) {
    const float* Wp = (const float*)d_in[4 + 4 * s];
    const float* bp = (const float*)d_in[5 + 4 * s];
    const float* Wd = (const float*)d_in[6 + 4 * s];
    const float* bd = (const float*)d_in[7 + 4 * s];
    const int n = ns[s], kk = kks[s];
    const int P = ((n + 31) / 32) * 32;

    // scores = sigmoid(relu(Wp * (g @ h) + bp))
    k_matvec_act<<<n, 256, 0, stream>>>(gcur, hcur, n, Wp, bp, 1, scores, nullptr);
    // top-k with exact tie order; also builds h_pool = h[idx] * values
    k_topk<<<1, 512, 0, stream>>>(scores, hcur, n, kk, idx, hpool);
    // binary adjacency -> padded f16 + transpose, then 2-hop via WMMA
    k_binpad<<<(P * P + 255) / 256, 256, 0, stream>>>(gcur, n, P, Gh, GTh);
    k_twohop_wmma<<<dim3(P / 16, P / 16), 32, 0, stream>>>(Gh, GTh, P, B2);
    // gather by idx, column-normalize (divide by row-sum of row j)
    k_rowsum<<<1, 512, 0, stream>>>(B2, P, idx, kk, S);
    float* gnext = (s & 1) ? gB : gA;
    k_gathernorm<<<(kk * kk + 255) / 256, 256, 0, stream>>>(B2, P, idx, S, kk, gnext);
    // down GCN: relu(sum_j g[i,j]*h_pool[j]*Wd[j] + bd) -> out[offs[s]:offs[s]+kk]
    k_prescale<<<(kk + 255) / 256, 256, 0, stream>>>(hpool, Wd, kk, xd);
    k_matvec_act<<<kk, 256, 0, stream>>>(gnext, xd, kk, nullptr, bd, 0, hcur, out + offs[s]);
    gcur = gnext;
  }
}
